// Merger_47639777247609
// MI455X (gfx1250) — matrix operand order repaired and checked by reading, not compiled, not run on hardware
//
#include <hip/hip_runtime.h>

#define HH 287
#define WW 352
#define NN (HH * WW)        // 101024
#define NCAM 4
#define NCH 13
#define NB 8
#define CIN1 56
#define CPAD 64             // padded channel count for both conv inputs
#define KCH3 18             // K chunks for 3x3 convs: 9 taps * 64 ch / 32

typedef __attribute__((ext_vector_type(16))) __bf16 v16bf;
typedef __attribute__((ext_vector_type(8)))  float  v8f;

union Frag16 {
    v16bf v;
    uint4 q[2];
    unsigned short us[16];
};

__device__ __forceinline__ unsigned short f2bf(float f) {
    unsigned int u = __builtin_bit_cast(unsigned int, f);
    u += 0x7FFFu + ((u >> 16) & 1u);          // round-to-nearest-even
    return (unsigned short)(u >> 16);
}
__device__ __forceinline__ unsigned pack2bf(float a, float b) {
    return (unsigned)f2bf(a) | ((unsigned)f2bf(b) << 16);
}

// gfx1250 async global->LDS copy (128b), tracked by ASYNCcnt
__device__ __forceinline__ void async_copy_b128(unsigned lds_off, const void* gptr) {
    unsigned long long ga = (unsigned long long)(size_t)gptr;
    asm volatile("global_load_async_to_lds_b128 %0, %1, off"
                 :: "v"(lds_off), "v"(ga) : "memory");
}
__device__ __forceinline__ void wait_asynccnt0() {
    asm volatile("s_wait_asynccnt 0x0" ::: "memory");
}

// ---------------------------------------------------------------------------
// M[t][s] = inv(T[t]) @ T[s]   (16 (t,s) pairs, one thread each)
// ---------------------------------------------------------------------------
__global__ void build_M_kernel(const float* __restrict__ T, float* __restrict__ Mout) {
    int tid = threadIdx.x;
    if (tid >= 16) return;
    int t = tid >> 2, s = tid & 3;
    float a[4][8];
    for (int i = 0; i < 4; ++i)
        for (int j = 0; j < 4; ++j) {
            a[i][j]     = T[t * 16 + i * 4 + j];
            a[i][4 + j] = (i == j) ? 1.0f : 0.0f;
        }
    for (int col = 0; col < 4; ++col) {
        int piv = col;
        float mx = fabsf(a[col][col]);
        for (int r = col + 1; r < 4; ++r) {
            float v = fabsf(a[r][col]);
            if (v > mx) { mx = v; piv = r; }
        }
        if (piv != col)
            for (int j = 0; j < 8; ++j) {
                float tmp = a[col][j]; a[col][j] = a[piv][j]; a[piv][j] = tmp;
            }
        float inv = 1.0f / a[col][col];
        for (int j = 0; j < 8; ++j) a[col][j] *= inv;
        for (int r = 0; r < 4; ++r) {
            if (r == col) continue;
            float f = a[r][col];
            for (int j = 0; j < 8; ++j) a[r][j] -= f * a[col][j];
        }
    }
    for (int i = 0; i < 4; ++i)
        for (int j = 0; j < 4; ++j) {
            float acc = 0.0f;
            for (int k = 0; k < 4; ++k)
                acc += a[i][4 + k] * T[s * 16 + k * 4 + j];
            Mout[(t * 4 + s) * 16 + i * 4 + j] = acc;
        }
}

// ---------------------------------------------------------------------------
// Pack OIHW fp32 weights into WMMA A-fragment layout (bf16 pairs).
// dst layout: [kc][mt][lane(32)][j(8)] as u32 (two packed bf16).
// A 16x32 bf16: m = lane&15, half = lane>>4,
//   pair j holds K = half*16 + j*2, +1 (contiguous, matches B).
// tapmode=1: K = tap*64 + c (3x3 convs);  tapmode=0: K = c (1x1).
// ---------------------------------------------------------------------------
__global__ void packA_kernel(const float* __restrict__ w, unsigned int* __restrict__ dst,
                             int MO, int MT, int CIreal, int TAPS, int KCH, int tapmode) {
    int idx = blockIdx.x * blockDim.x + threadIdx.x;
    int total = KCH * MT * 32 * 8;
    if (idx >= total) return;
    int j    = idx & 7;
    int lane = (idx >> 3) & 31;
    int rest = idx >> 8;
    int mt   = rest % MT;
    int kc   = rest / MT;
    int m    = mt * 16 + (lane & 15);
    int half = lane >> 4;
    int kpb  = half * 16 + j * 2;
    unsigned short p[2];
    for (int q = 0; q < 2; ++q) {
        int k = kc * 32 + kpb + q;
        float val = 0.0f;
        if (tapmode) {
            int tap = k >> 6, c = k & 63;
            if (m < MO && c < CIreal)
                val = w[((size_t)m * CIreal + c) * TAPS + tap];
        } else {
            if (m < MO && k < CIreal)
                val = w[(size_t)m * CIreal + k];
        }
        p[q] = f2bf(val);
    }
    dst[idx] = (unsigned int)p[0] | ((unsigned int)p[1] << 16);
}

// ---------------------------------------------------------------------------
// Zero-fill (128b granularity)
// ---------------------------------------------------------------------------
__global__ void zero4_kernel(uint4* __restrict__ p, size_t nquads) {
    size_t i = (size_t)blockIdx.x * blockDim.x + threadIdx.x;
    if (i < nquads) p[i] = make_uint4(0u, 0u, 0u, 0u);
}

// ---------------------------------------------------------------------------
// Scatter-build the pixel-major input X[b][N][64] (bf16) for target camera t.
// Channels s*14 .. s*14+13 are 14 contiguous u16 -> 7 aligned u32 stores.
// grid.z = s*8 + b ; one thread per source pixel.
// ---------------------------------------------------------------------------
__global__ void scatter_kernel(const float* __restrict__ feat,   // [B][CAM][13][N]
                               const float* __restrict__ xyz,    // [B][CAM][3][N]
                               const float* __restrict__ Mmat,   // [4][4][16]
                               const float* __restrict__ Kmat,   // [CAM][9]
                               unsigned int* __restrict__ X,     // [B][N][64] bf16 as u32 pairs
                               int t) {
    int n = blockIdx.x * blockDim.x + threadIdx.x;
    if (n >= NN) return;
    int sz = blockIdx.z;
    int s = sz >> 3, b = sz & 7;
    const float* M = Mmat + (t * 4 + s) * 16;
    size_t xb = ((size_t)(b * NCAM + s) * 3) * NN + n;
    float x = xyz[xb], y = xyz[xb + NN], z = xyz[xb + 2 * (size_t)NN];
    float px = M[0] * x + M[1] * y + M[2]  * z + M[3];
    float py = M[4] * x + M[5] * y + M[6]  * z + M[7];
    float pz = M[8] * x + M[9] * y + M[10] * z + M[11];
    int dst;
    if (s == t) {
        dst = n;
    } else {
        float xn = px / pz, yn = py / pz;
        const float* Kt = Kmat + t * 9;
        float xi = rintf(Kt[0] * xn + Kt[1] * yn + Kt[2]);
        float yi = rintf(Kt[3] * xn + Kt[4] * yn + Kt[5]);
        bool valid = (xi > 0.0f) && (xi < (float)WW) && (yi > 0.0f) && (yi < (float)HH)
                     && isfinite(xi) && isfinite(yi);
        if (!valid) return;
        int ix = (int)xi, iy = (int)yi;
        dst = (HH - 1 - iy) * WW + ix;      // vertical flip of projected image
    }
    size_t fb = ((size_t)(b * NCAM + s) * NCH) * NN + n;
    float f[14];
    #pragma unroll
    for (int c = 0; c < NCH; ++c)
        f[c] = 2.0f * feat[fb + (size_t)c * NN] - 1.0f;
    f[13] = pz;
    // 7 u32 words at channel offset s*14 of pixel dst
    unsigned int* xp = X + (((size_t)b * NN + dst) * CPAD + s * 14) / 2;
    #pragma unroll
    for (int i = 0; i < 7; ++i)
        xp[i] = pack2bf(f[2 * i], f[2 * i + 1]);
}

// ---------------------------------------------------------------------------
// Implicit-GEMM 3x3 conv (64->64 padded) + BN + ReLU, WMMA bf16.
// Block = 256 threads (8 waves), one image row, 128 pixels.
// Wave: 16-pixel column x 64 out channels (4 M-tiles), K = tap*64+c, 18 chunks.
// Halo tile [3][130][64] staged in LDS via async b128 copies.
// ---------------------------------------------------------------------------
__global__ __launch_bounds__(256) void conv3x3_bn_relu_wmma(
        const unsigned short* __restrict__ Xin,    // [B][N][64] bf16
        const unsigned int*  __restrict__ Apack,   // [18][4][32][8]
        const float* __restrict__ gamma,
        const float* __restrict__ beta,
        unsigned short* __restrict__ Yout) {       // [B][N][64] bf16
    __shared__ unsigned short smem[3 * 130 * CPAD];   // 49920 B
    const int b  = blockIdx.z;
    const int h  = blockIdx.y;
    const int w0 = blockIdx.x * 128;
    const int tid = threadIdx.x;

    // halo load: 3 rows x 130 pixels x 128B, 16B chunks (8 per pixel slot)
    {
        const unsigned lds0 = (unsigned)(size_t)(&smem[0]);
        const int nchunks = 3 * 130 * 8;               // 3120
        for (int i = tid; i < nchunks; i += 256) {
            int slot = i >> 3;                          // 0..389
            int sub  = i & 7;                           // 16B sub-chunk
            int r    = slot / 130;
            int l    = slot - r * 130;
            int gh   = h + r - 1;
            int gw   = w0 - 1 + l;
            if (gh >= 0 && gh < HH && gw >= 0 && gw < WW) {
                const void* g = Xin + (((size_t)b * NN + (size_t)gh * WW + gw) * CPAD + sub * 8);
                async_copy_b128(lds0 + (unsigned)i * 16u, g);
            } else {
                *reinterpret_cast<uint4*>(&smem[(size_t)i * 8]) = make_uint4(0u, 0u, 0u, 0u);
            }
        }
    }
    wait_asynccnt0();
    __syncthreads();

    const int lane = tid & 31;
    const int wave = tid >> 5;
    const int half = lane >> 4;
    const int scol = wave * 16 + (lane & 15);   // 0..127 within tile

    v8f acc[4];
    #pragma unroll
    for (int i = 0; i < 4; ++i)
        #pragma unroll
        for (int r = 0; r < 8; ++r) acc[i][r] = 0.0f;

    #pragma unroll 2
    for (int kc = 0; kc < KCH3; ++kc) {
        // B fragment: 16 consecutive K = 16 consecutive channels at one tap
        int kb  = kc * 32 + half * 16;
        int tap = kb >> 6;
        int c0  = kb & 63;
        int dy  = tap / 3, dx = tap - 3 * dy;
        Frag16 bfr;
        const uint4* bp = reinterpret_cast<const uint4*>(
            &smem[((size_t)(dy * 130 + scol + dx)) * CPAD + c0]);
        bfr.q[0] = bp[0];
        bfr.q[1] = bp[1];
        #pragma unroll
        for (int mt = 0; mt < 4; ++mt) {
            Frag16 afr;
            const uint4* ap = reinterpret_cast<const uint4*>(Apack)
                              + ((size_t)(kc * 4 + mt) * 32 + lane) * 2;
            afr.q[0] = ap[0];
            afr.q[1] = ap[1];
            acc[mt] = __builtin_amdgcn_wmma_f32_16x16x32_bf16(
                false, afr.v, false, bfr.v, (short)0, acc[mt], false, false);
        }
    }

    // fused BN (eval) + ReLU, packed b128 stores (8 consecutive channels / mt)
    const int w = w0 + scol;
    if (w < WW) {
        const float bnrs = rsqrtf(1.0f + 1e-5f);
        size_t pbase = ((size_t)b * NN + (size_t)h * WW + w) * CPAD;
        #pragma unroll
        for (int mt = 0; mt < 4; ++mt) {
            int ch0 = mt * 16 + 8 * half;
            float v[8];
            #pragma unroll
            for (int r = 0; r < 8; ++r) {
                int ch = ch0 + r;
                float val = acc[mt][r] * (gamma[ch] * bnrs) + beta[ch];
                v[r] = fmaxf(val, 0.0f);
            }
            uint4 q = make_uint4(pack2bf(v[0], v[1]), pack2bf(v[2], v[3]),
                                 pack2bf(v[4], v[5]), pack2bf(v[6], v[7]));
            *reinterpret_cast<uint4*>(Yout + pbase + ch0) = q;
        }
    }
}

// ---------------------------------------------------------------------------
// 1x1 conv (64 -> 13, M padded to 16) + bias, fp32 output, WMMA bf16.
// B fragment = two b128 loads from pixel-major Y2 (index clamped, no predication).
// ---------------------------------------------------------------------------
__global__ __launch_bounds__(256) void conv1x1_bias_wmma(
        const unsigned short* __restrict__ Y2,    // [B][N][64] bf16
        const unsigned int*  __restrict__ A3,     // [2][1][32][8]
        const float* __restrict__ bias3,
        float* __restrict__ Out,                  // [CAM][B][13][N]
        int t) {
    const int lane = threadIdx.x & 31;
    const int wave = threadIdx.x >> 5;
    const int half = lane >> 4;
    const int b = blockIdx.z;
    const int n = blockIdx.x * 128 + wave * 16 + (lane & 15);
    const int nc = (n < NN) ? n : (NN - 1);      // clamp: loads always legal

    v8f acc;
    #pragma unroll
    for (int r = 0; r < 8; ++r) acc[r] = 0.0f;

    const unsigned short* ybase = Y2 + ((size_t)b * NN + nc) * CPAD;
    #pragma unroll
    for (int kc = 0; kc < 2; ++kc) {
        Frag16 afr, bfr;
        const uint4* ap = reinterpret_cast<const uint4*>(A3) + ((size_t)kc * 32 + lane) * 2;
        afr.q[0] = ap[0];
        afr.q[1] = ap[1];
        const uint4* bp = reinterpret_cast<const uint4*>(ybase + kc * 32 + half * 16);
        bfr.q[0] = bp[0];
        bfr.q[1] = bp[1];
        acc = __builtin_amdgcn_wmma_f32_16x16x32_bf16(
            false, afr.v, false, bfr.v, (short)0, acc, false, false);
    }

    if (n < NN) {
        #pragma unroll
        for (int r = 0; r < 8; ++r) {
            int m = r + 8 * half;
            if (m < NCH)
                Out[((size_t)(t * NB + b) * NCH + m) * NN + n] = acc[r] + bias3[m];
        }
    }
}

// ---------------------------------------------------------------------------
// Host-side launch sequence
// ---------------------------------------------------------------------------
extern "C" void kernel_launch(void* const* d_in, const int* in_sizes, int n_in,
                              void* d_out, int out_size, void* d_ws, size_t ws_size,
                              hipStream_t stream) {
    (void)in_sizes; (void)n_in; (void)out_size; (void)ws_size;
    const float* feat = (const float*)d_in[0];   // (8,4,13,H,W)
    const float* xyz  = (const float*)d_in[1];   // (8,4,3,H,W)
    const float* T    = (const float*)d_in[2];   // (4,4,4)
    const float* Km   = (const float*)d_in[3];   // (4,3,3)
    const float* w1   = (const float*)d_in[4];   // (64,56,3,3)
    const float* g1   = (const float*)d_in[5];
    const float* b1   = (const float*)d_in[6];
    const float* w2   = (const float*)d_in[7];   // (64,64,3,3)
    const float* g2   = (const float*)d_in[8];
    const float* b2   = (const float*)d_in[9];
    const float* w3   = (const float*)d_in[10];  // (13,64,1,1)
    const float* b3   = (const float*)d_in[11];
    float* out = (float*)d_out;

    char* ws = (char*)d_ws;
    size_t off = 0;
    auto take = [&](size_t bytes) {
        off = (off + 255) & ~(size_t)255;
        char* p = ws + off;
        off += bytes;
        return p;
    };
    float*          Mws = (float*)         take(16 * 16 * sizeof(float));
    unsigned int*   A1  = (unsigned int*)  take((size_t)KCH3 * 4 * 32 * 8 * 4);
    unsigned int*   A2  = (unsigned int*)  take((size_t)KCH3 * 4 * 32 * 8 * 4);
    unsigned int*   A3  = (unsigned int*)  take((size_t)2    * 1 * 32 * 8 * 4);
    unsigned short* X   = (unsigned short*)take((size_t)NB * NN * CPAD * 2);
    unsigned short* Y1  = (unsigned short*)take((size_t)NB * NN * CPAD * 2);
    unsigned short* Y2  = (unsigned short*)take((size_t)NB * NN * CPAD * 2);

    // one-time prep
    build_M_kernel<<<1, 16, 0, stream>>>(T, Mws);
    {
        int tot3x3 = KCH3 * 4 * 32 * 8;
        packA_kernel<<<(tot3x3 + 255) / 256, 256, 0, stream>>>(w1, A1, 64, 4, CIN1, 9, KCH3, 1);
        packA_kernel<<<(tot3x3 + 255) / 256, 256, 0, stream>>>(w2, A2, 64, 4, CPAD, 9, KCH3, 1);
        int tot1x1 = 2 * 1 * 32 * 8;
        packA_kernel<<<(tot1x1 + 255) / 256, 256, 0, stream>>>(w3, A3, NCH, 1, CPAD, 1, 2, 0);
    }

    const size_t xquads = ((size_t)NB * NN * CPAD * 2) / 16;
    dim3 convGrid((WW + 127) / 128, HH, NB);
    dim3 c3Grid((NN + 127) / 128, 1, NB);

    for (int t = 0; t < NCAM; ++t) {
        zero4_kernel<<<(unsigned)((xquads + 255) / 256), 256, 0, stream>>>((uint4*)X, xquads);
        dim3 sg((NN + 255) / 256, 1, NCAM * NB);
        scatter_kernel<<<sg, 256, 0, stream>>>(feat, xyz, Mws, Km, (unsigned int*)X, t);
        conv3x3_bn_relu_wmma<<<convGrid, 256, 0, stream>>>(X,  A1, g1, b1, Y1);
        conv3x3_bn_relu_wmma<<<convGrid, 256, 0, stream>>>(Y1, A2, g2, b2, Y2);
        conv1x1_bias_wmma<<<c3Grid, 256, 0, stream>>>(Y2, A3, b3, out, t);
    }
}
